// RNNDecoder_3212635538035
// MI455X (gfx1250) — compile-verified, matrix operand order checked by reading, not run on hardware
//
#include <hip/hip_runtime.h>
#include <hip/hip_bf16.h>

// ---------------------------------------------------------------------------
// Bidirectional 2-layer GRU for MI455X (gfx1250, wave32, WMMA).
//   B=64 batch, T=512 time, H=256 hidden, Cin=512 both layers.
// All GEMMs in bf16 via v_wmma_f32_16x16x32_bf16, f32 accumulate.
// Scratch layout (requires ws_size >= ~165 MB):
//   seqA  bf16 (B,T,512)      layer-0 input (transposed x)
//   seqB  bf16 (B,T,512)      layer-0 output / layer-1 input
//   XP    bf16 (B*T,1536)     input projections, fwd cols 0..767, rev 768..1535
//   packed bf16 weights
// ---------------------------------------------------------------------------

typedef __attribute__((ext_vector_type(16))) __bf16 v16bf;
typedef __attribute__((ext_vector_type(8)))  __bf16 v8bf;
typedef __attribute__((ext_vector_type(8)))  float  v8f;

constexpr int kB   = 64;
constexpr int kT   = 512;
constexpr int kH   = 256;
constexpr int kCIN = 512;
constexpr int kG3  = 768;    // 3*H
constexpr int kNPK = 1536;   // fwd + rev gate columns packed
constexpr int kMT  = 64;     // GEMM block M tile (4 x 16 per wave)

union FragU { v16bf v; v8bf h[2]; };

// Load a 16x32 bf16 fragment (A-layout, ISA 7.12.2) from a row-major matrix.
// lane 0-15 : row = lane,    K = kofs+0..7  (VGPR0-3) and kofs+16..23 (VGPR4-7)
// lane16-31 : row = lane-16, K = kofs+8..15 (VGPR0-3) and kofs+24..31 (VGPR4-7)
// The B-matrix (32x16, K x N) uses the mirrored layout: pass base = &W[n0*ld]
// so "row" indexes the output column n0+lane (W stored as N x K row-major).
__device__ __forceinline__ v16bf load_frag_rowmajor(const __bf16* __restrict__ base,
                                                    int ld, int kofs) {
  const int lane = threadIdx.x & 31;
  const int row  = lane & 15;
  const int kb   = kofs + ((lane >> 4) << 3);
  const __bf16* p = base + (size_t)row * ld + kb;
  FragU u;
  u.h[0] = *(const v8bf*)(p);        // K = kb   .. kb+7   (16B -> b128 load)
  u.h[1] = *(const v8bf*)(p + 16);   // K = kb+16.. kb+23
  return u.v;
}

// ---------------------------------------------------------------------------
// fp32 -> bf16 weight packing
// ---------------------------------------------------------------------------
__global__ void cvt_f32_bf16(const float* __restrict__ src, __bf16* __restrict__ dst, int n) {
  int i = blockIdx.x * blockDim.x + threadIdx.x;
  if (i < n) dst[i] = (__bf16)src[i];
}

// ---------------------------------------------------------------------------
// x (b, c, t) fp32  ->  seq (b, t, c) bf16   (LDS-tiled 32x32 transpose)
// ---------------------------------------------------------------------------
__global__ __launch_bounds__(256)
void transpose_cast_kernel(const float* __restrict__ X, __bf16* __restrict__ S) {
  __shared__ __bf16 tile[32][34];          // padded to dodge bank conflicts
  const int b  = blockIdx.z;
  const int c0 = blockIdx.y * 32;
  const int t0 = blockIdx.x * 32;
  const int tx = threadIdx.x & 31;
  const int ty = threadIdx.x >> 5;         // 0..7
  for (int r = 0; r < 4; ++r) {
    const int c = c0 + ty + r * 8;
    tile[ty + r * 8][tx] = (__bf16)X[((size_t)(b * 512 + c)) * 512 + t0 + tx];
  }
  __syncthreads();
  for (int r = 0; r < 4; ++r) {
    const int t = t0 + ty + r * 8;
    S[((size_t)(b * 512 + t)) * 512 + c0 + tx] = tile[tx][ty + r * 8];
  }
}

// ---------------------------------------------------------------------------
// Input projection GEMM: XP[m, n] = seq[m, :] . Wpk[n, :] + b_ih[n]
//   M = B*T = 32768, N = 1536 (fwd 0..767 | rev 768..1535), K = 512.
// Block = 8 waves, block tile M=64 x N=128. Each wave owns one 16-wide N
// column tile and 4 M sub-tiles: one B fragment feeds 4 WMMAs (4x less
// global B traffic per FLOP than 1x1 blocking). 64x512 A tile staged in LDS
// (one contiguous 64 KB global block since ld == 512) via async-to-LDS.
// ---------------------------------------------------------------------------
__global__ __launch_bounds__(256)
void wih_gemm_kernel(const __bf16* __restrict__ A, const __bf16* __restrict__ W,
                     const float* __restrict__ bias_f, const float* __restrict__ bias_r,
                     __bf16* __restrict__ XP) {
  __shared__ __bf16 a_tile[kMT * kCIN];                    // 64 KB
  const int m0 = blockIdx.y * kMT;

  // stage A tile: contiguous 64 KB -> LDS
  {
    // prefetch next M-tile's A rows into L2 while we work on this one
    __builtin_prefetch(A + (size_t)(m0 + kMT) * kCIN + threadIdx.x * 128, 0, 1);
#if __has_builtin(__builtin_amdgcn_global_load_async_to_lds_b128) && \
    __has_builtin(__builtin_amdgcn_s_wait_asynccnt)
    // builtin prototype wants (global int4-vec*, lds int4-vec*, imm, imm)
    typedef int v4i_g __attribute__((vector_size(16)));
    __attribute__((address_space(1))) v4i_g* g =
        (__attribute__((address_space(1))) v4i_g*)(A + (size_t)m0 * kCIN);
    __attribute__((address_space(3))) v4i_g* l =
        (__attribute__((address_space(3))) v4i_g*)a_tile;
    for (int i = threadIdx.x; i < (kMT * kCIN * 2) / 16; i += blockDim.x)
      __builtin_amdgcn_global_load_async_to_lds_b128(g + i, l + i, 0, 0);
    __builtin_amdgcn_s_wait_asynccnt(0);
#else
    const uint4* src = (const uint4*)(A + (size_t)m0 * kCIN);
    uint4* dst = (uint4*)a_tile;
    for (int i = threadIdx.x; i < (kMT * kCIN) / 8; i += blockDim.x) dst[i] = src[i];
#endif
  }
  __syncthreads();

  const int wave  = threadIdx.x >> 5;
  const int lane  = threadIdx.x & 31;
  const int ncol0 = blockIdx.x * 128 + wave * 16;          // wave's N tile base
  const __bf16* wbase = W + (size_t)ncol0 * kCIN;          // W is (N x K) row-major

  v8f acc[4] = {{}, {}, {}, {}};
  for (int kk = 0; kk < 16; ++kk) {
    const v16bf bfrag = load_frag_rowmajor(wbase, kCIN, kk * 32); // global_load_b128 x2
    for (int mt = 0; mt < 4; ++mt) {
      const v16bf afrag =
          load_frag_rowmajor(a_tile + (size_t)mt * 16 * kCIN, kCIN, kk * 32); // ds x2
      acc[mt] = __builtin_amdgcn_wmma_f32_16x16x32_bf16(false, afrag, false, bfrag,
                                                        (short)0, acc[mt], false, false);
    }
  }

  // epilogue: + b_ih, cast to bf16, scatter per C-fragment layout
  const int col  = ncol0 + (lane & 15);
  const float bias = (col < kG3) ? bias_f[col] : bias_r[col - kG3];
  const int mo = (lane >> 4) << 3;
  for (int mt = 0; mt < 4; ++mt) {
    __bf16* out = XP + (size_t)(m0 + mt * 16) * kNPK + col;
    for (int v = 0; v < 8; ++v) {
      const int m = mo + v;
      out[(size_t)m * kNPK] = (__bf16)(acc[mt][v] + bias);
    }
  }
}

// ---------------------------------------------------------------------------
// GRU recurrence. Grid = 8 blocks: (dir = bx&1, batch tile of 16 = bx>>1).
// h (16 x 256 bf16) lives in LDS. Per timestep each wave computes 6 gh tiles
// (gates r,z,n x 2 column sub-tiles) = 48 WMMAs, then the gate nonlinearity.
// W_hh (384 KB/dir) streams from the 192 MB L2 every step (cannot fit LDS).
// ---------------------------------------------------------------------------
template <bool FINAL>
__global__ __launch_bounds__(256)
void gru_rec_kernel(const __bf16* __restrict__ XP, const __bf16* __restrict__ WHH,
                    const float* __restrict__ bhh_f, const float* __restrict__ bhh_r,
                    __bf16* __restrict__ Yb, float* __restrict__ Yf) {
  __shared__ __bf16 hbuf[16 * kH];                         // 8 KB
  const int dir  = blockIdx.x & 1;
  const int bt   = blockIdx.x >> 1;                        // 0..3
  const int wave = threadIdx.x >> 5;
  const int lane = threadIdx.x & 31;
  const int jc   = lane & 15;
  const int mo   = (lane >> 4) << 3;

  for (int i = threadIdx.x; i < 16 * kH; i += blockDim.x) hbuf[i] = (__bf16)0.0f;
  __syncthreads();

  const __bf16* Wd  = WHH + (size_t)dir * kG3 * kH;        // (768 x 256) row-major
  const float*  bhh = dir ? bhh_r : bhh_f;

  // hidden-unit biases are loop-invariant: hoist out of the t loop
  float bg[2][3];
  for (int s = 0; s < 2; ++s) {
    const int j = wave * 32 + s * 16 + jc;
    bg[s][0] = bhh[j];
    bg[s][1] = bhh[kH + j];
    bg[s][2] = bhh[2 * kH + j];
  }

  for (int t = 0; t < kT; ++t) {
    const int tm = dir ? (kT - 1 - t) : t;

    // A fragments of h: load once, reuse across all 6 output tiles
    v16bf afr[8];
    for (int kk = 0; kk < 8; ++kk)
      afr[kk] = load_frag_rowmajor(hbuf, kH, kk * 32);     // ds_load_b128 x2

    v8f acc[2][3];
    for (int s = 0; s < 2; ++s)
      for (int g = 0; g < 3; ++g) {
        const int ncol = g * kH + wave * 32 + s * 16;
        const __bf16* wb = Wd + (size_t)ncol * kH;
        v8f c = {};
        for (int kk = 0; kk < 8; ++kk) {
          const v16bf bfrag = load_frag_rowmajor(wb, kH, kk * 32); // L2-resident
          c = __builtin_amdgcn_wmma_f32_16x16x32_bf16(false, afr[kk], false, bfrag,
                                                      (short)0, c, false, false);
        }
        acc[s][g] = c;
      }

    // gates: r = sig(xr+hr), z = sig(xz+hz), n = tanh(xn + r*hn)
    float hnew[2][8];
    for (int s = 0; s < 2; ++s) {
      const int j = wave * 32 + s * 16 + jc;
      for (int v = 0; v < 8; ++v) {
        const int m = mo + v;
        const int b = bt * 16 + m;
        const size_t xrow = ((size_t)b * kT + tm) * kNPK + (size_t)dir * kG3;
        const float xr = (float)XP[xrow + j];
        const float xz = (float)XP[xrow + kH + j];
        const float xn = (float)XP[xrow + 2 * kH + j];
        const float hold = (float)hbuf[m * kH + j];
        const float r  = 1.0f / (1.0f + __expf(-(xr + acc[s][0][v] + bg[s][0])));
        const float z  = 1.0f / (1.0f + __expf(-(xz + acc[s][1][v] + bg[s][1])));
        const float nv = xn + r * (acc[s][2][v] + bg[s][2]);
        const float n  = 2.0f / (1.0f + __expf(-2.0f * nv)) - 1.0f;   // tanh
        hnew[s][v] = (1.0f - z) * n + z * hold;
      }
    }

    __syncthreads();   // everyone done reading h for this step
    for (int s = 0; s < 2; ++s) {
      const int j = wave * 32 + s * 16 + jc;
      for (int v = 0; v < 8; ++v) {
        const int m = mo + v;
        const int b = bt * 16 + m;
        hbuf[m * kH + j] = (__bf16)hnew[s][v];
        const size_t o = ((size_t)b * kT + tm) * (2 * kH) + (size_t)dir * kH + j;
        if (FINAL) Yf[o] = hnew[s][v];
        else       Yb[o] = (__bf16)hnew[s][v];
      }
    }
    __syncthreads();   // h updated before next step's reads
  }
}

// ---------------------------------------------------------------------------
extern "C" void kernel_launch(void* const* d_in, const int* in_sizes, int n_in,
                              void* d_out, int out_size, void* d_ws, size_t ws_size,
                              hipStream_t stream) {
  (void)in_sizes; (void)n_in; (void)out_size; (void)ws_size;

  const float* x         = (const float*)d_in[0];
  const float* w_ih_l0f  = (const float*)d_in[1];
  const float* w_hh_l0f  = (const float*)d_in[2];
  const float* b_ih_l0f  = (const float*)d_in[3];
  const float* b_hh_l0f  = (const float*)d_in[4];
  const float* w_ih_l0r  = (const float*)d_in[5];
  const float* w_hh_l0r  = (const float*)d_in[6];
  const float* b_ih_l0r  = (const float*)d_in[7];
  const float* b_hh_l0r  = (const float*)d_in[8];
  const float* w_ih_l1f  = (const float*)d_in[9];
  const float* w_hh_l1f  = (const float*)d_in[10];
  const float* b_ih_l1f  = (const float*)d_in[11];
  const float* b_hh_l1f  = (const float*)d_in[12];
  const float* w_ih_l1r  = (const float*)d_in[13];
  const float* w_hh_l1r  = (const float*)d_in[14];
  const float* b_ih_l1r  = (const float*)d_in[15];
  const float* b_hh_l1r  = (const float*)d_in[16];

  const size_t SEQ_E = (size_t)kB * kT * 512;     // 16.7M bf16
  const size_t XP_E  = (size_t)kB * kT * kNPK;    // 50.3M bf16
  char* ws = (char*)d_ws;
  __bf16* seqA = (__bf16*)(ws);
  __bf16* seqB = (__bf16*)(ws + SEQ_E * 2);
  __bf16* XP   = (__bf16*)(ws + SEQ_E * 4);
  __bf16* wih0 = (__bf16*)(ws + SEQ_E * 4 + XP_E * 2);
  __bf16* wih1 = wih0 + (size_t)kNPK * kCIN;
  __bf16* whh0 = wih1 + (size_t)kNPK * kCIN;
  __bf16* whh1 = whh0 + (size_t)2 * kG3 * kH;

  const int nIH = kG3 * kCIN;   // 393216
  const int nHH = kG3 * kH;     // 196608
  const dim3 cb(256);
  cvt_f32_bf16<<<(nIH + 255) / 256, cb, 0, stream>>>(w_ih_l0f, wih0,        nIH);
  cvt_f32_bf16<<<(nIH + 255) / 256, cb, 0, stream>>>(w_ih_l0r, wih0 + nIH,  nIH);
  cvt_f32_bf16<<<(nIH + 255) / 256, cb, 0, stream>>>(w_ih_l1f, wih1,        nIH);
  cvt_f32_bf16<<<(nIH + 255) / 256, cb, 0, stream>>>(w_ih_l1r, wih1 + nIH,  nIH);
  cvt_f32_bf16<<<(nHH + 255) / 256, cb, 0, stream>>>(w_hh_l0f, whh0,        nHH);
  cvt_f32_bf16<<<(nHH + 255) / 256, cb, 0, stream>>>(w_hh_l0r, whh0 + nHH,  nHH);
  cvt_f32_bf16<<<(nHH + 255) / 256, cb, 0, stream>>>(w_hh_l1f, whh1,        nHH);
  cvt_f32_bf16<<<(nHH + 255) / 256, cb, 0, stream>>>(w_hh_l1r, whh1 + nHH,  nHH);

  // (b,c,t) -> (b,t,c) bf16
  transpose_cast_kernel<<<dim3(16, 16, kB), dim3(256), 0, stream>>>(x, seqA);

  // layer 0
  wih_gemm_kernel<<<dim3(kNPK / 128, (kB * kT) / kMT), dim3(256), 0, stream>>>(
      seqA, wih0, b_ih_l0f, b_ih_l0r, XP);
  gru_rec_kernel<false><<<dim3(8), dim3(256), 0, stream>>>(
      XP, whh0, b_hh_l0f, b_hh_l0r, seqB, nullptr);

  // layer 1
  wih_gemm_kernel<<<dim3(kNPK / 128, (kB * kT) / kMT), dim3(256), 0, stream>>>(
      seqB, wih1, b_ih_l1f, b_ih_l1r, XP);
  gru_rec_kernel<true><<<dim3(8), dim3(256), 0, stream>>>(
      XP, whh1, b_hh_l1f, b_hh_l1r, nullptr, (float*)d_out);
}
